// DETR_63419487092880
// MI455X (gfx1250) — matrix operand order repaired and checked by reading, not compiled, not run on hardware
//
#include <hip/hip_runtime.h>
#include <hip/hip_bf16.h>

typedef __attribute__((ext_vector_type(16))) __bf16 v16bf;
typedef __attribute__((ext_vector_type(8)))  float  v8f;

#define ROWS 5120      // B*N = 4*1280
#define NANCH 1280
#define HID 256

// ---------------------------------------------------------------------------
// Weight pre-transpose + fp32 -> bf16 convert: in (N x K) row-major fp32,
// out (K x N) bf16 so WMMA B-fragments are contiguous 32B loads.
// ---------------------------------------------------------------------------
__global__ void __launch_bounds__(256)
convert_wt_kernel(const float* __restrict__ W, unsigned short* __restrict__ Bt,
                  int N, int K) {
    int i = blockIdx.x * blockDim.x + threadIdx.x;
    if (i >= N * K) return;
    int n = i / K, k = i - n * K;
    __bf16 v = (__bf16)W[i];
    Bt[(size_t)k * N + n] = __builtin_bit_cast(unsigned short, v);
}

// ---------------------------------------------------------------------------
// Bilinear gather: one block per token (row = b*N + n), 256 lanes = channels.
// feat: (B,256,H,W) fp32, anchors: (rows,2), optional off: (rows,2).
// px = ax*(W-1)/(s*W) and s*W == 512 for every level -> scale = (W-1)/512.
// ---------------------------------------------------------------------------
__global__ void __launch_bounds__(256)
sample_kernel(const float* __restrict__ feat, const float* __restrict__ anchors,
              const float* __restrict__ off, float* __restrict__ out,
              int HW, float scale, int outStride) {
    int row = blockIdx.x;
    int c   = threadIdx.x;
    int b   = row / NANCH;
    float ax = anchors[row * 2 + 0], ay = anchors[row * 2 + 1];
    if (off) { ax += off[row * 2 + 0]; ay += off[row * 2 + 1]; }
    float px = ax * scale, py = ay * scale;
    float x0f = floorf(px), y0f = floorf(py);
    float fx = px - x0f, fy = py - y0f;
    int x0 = (int)x0f, y0 = (int)y0f;
    const float* fb = feat + ((size_t)b * HID + c) * (size_t)(HW * HW);
    float wx[2] = {1.f - fx, fx}, wy[2] = {1.f - fy, fy};
    float acc = 0.f;
#pragma unroll
    for (int dy = 0; dy < 2; ++dy) {
        int yi = y0 + dy;
        if (yi < 0 || yi >= HW) continue;
#pragma unroll
        for (int dx = 0; dx < 2; ++dx) {
            int xi = x0 + dx;
            if (xi < 0 || xi >= HW) continue;
            acc += wy[dy] * wx[dx] * fb[yi * HW + xi];
        }
    }
    out[(size_t)row * outStride + c] = acc;
}

// ---------------------------------------------------------------------------
// Fragment loader for the WMMA GEMM.
// A fragment: half j -> k = k0 + (j<8?0:16) + 8*kg + (j&7), cvt fp32->bf16.
// B fragments: 4 adjacent 16x16 n-tiles, per lane one bf16 per k row (stride N) and tile
// from the pre-transposed (K x N) bf16 weights.
// ---------------------------------------------------------------------------
__device__ __forceinline__ void
load_frag(const float* __restrict__ arow, const unsigned short* __restrict__ brow,
          int N, int k0, int kg, v16bf& a, v16bf* b) {
    const float4* p0 = reinterpret_cast<const float4*>(arow + k0 + 8 * kg);
    const float4* p1 = reinterpret_cast<const float4*>(arow + k0 + 16 + 8 * kg);
    float4 a0 = p0[0], a1 = p0[1], a2 = p1[0], a3 = p1[1];
    float af[16] = {a0.x, a0.y, a0.z, a0.w, a1.x, a1.y, a1.z, a1.w,
                    a2.x, a2.y, a2.z, a2.w, a3.x, a3.y, a3.z, a3.w};
#pragma unroll
    for (int j = 0; j < 16; ++j) a[j] = (__bf16)af[j];
    const unsigned short* bp = brow + (size_t)k0 * N;
#pragma unroll
    for (int t = 0; t < 4; ++t)
#pragma unroll
        for (int j = 0; j < 16; ++j)
            b[t][j] = __builtin_bit_cast(__bf16, bp[(size_t)((j & 7) + ((j >> 3) << 4)) * N + 16 * t]);
}

// ---------------------------------------------------------------------------
// WMMA GEMM, 1x4 n-tile blocking + 2-stage software pipeline: each wave
// computes a 16x64 slice of C = act(A(MxK,fp32) * Bt(KxN,bf16) + bias).
// Loads for k+32 are issued before the wmmas for k so VMEM latency overlaps
// matrix math. Requires N % 64 == 0 and K % 64 == 0 (all call sites comply).
// ---------------------------------------------------------------------------
__global__ void __launch_bounds__(128)
gemm_bf16_wmma_kernel(const float* __restrict__ A,
                      const unsigned short* __restrict__ Bt,
                      const float* __restrict__ bias, float* __restrict__ C,
                      int M, int N, int K, int relu) {
    int wave = (int)((blockIdx.x * blockDim.x + threadIdx.x) >> 5);
    int lane = threadIdx.x & 31;
    int nblk  = N >> 6;                  // 16x64 output blocks per row band
    int total = (M >> 4) * nblk;
    if (wave >= total) return;           // wave-uniform: EXEC all-1s for WMMA
    int mT = wave / nblk, nB = wave - mT * nblk;
    int kg = lane >> 4;                  // A K-group (0/1)
    const float* arow = A + (size_t)(mT * 16 + (lane & 15)) * K;
    // per-lane B base: lane -> column n (lane&15), K rows 8*kg (+j&7, +16 for j>=8)
    int klofs = (lane >> 4) << 3;
    const unsigned short* brow = Bt + (size_t)klofs * N + (nB << 6) + (lane & 15);

    v8f acc[4];
#pragma unroll
    for (int t = 0; t < 4; ++t) acc[t] = (v8f){0.f,0.f,0.f,0.f,0.f,0.f,0.f,0.f};

    v16bf fa, fb[4], ga, gb[4];
    load_frag(arow, brow, N, 0, kg, fa, fb);
    for (int k0 = 0; k0 < K; k0 += 64) {
        // stage 1: prefetch k0+32, then consume k0
        load_frag(arow, brow, N, k0 + 32, kg, ga, gb);
        acc[0] = __builtin_amdgcn_wmma_f32_16x16x32_bf16(false, fa, false, fb[0],
                                                         (short)0, acc[0], false, false);
        acc[1] = __builtin_amdgcn_wmma_f32_16x16x32_bf16(false, fa, false, fb[1],
                                                         (short)0, acc[1], false, false);
        acc[2] = __builtin_amdgcn_wmma_f32_16x16x32_bf16(false, fa, false, fb[2],
                                                         (short)0, acc[2], false, false);
        acc[3] = __builtin_amdgcn_wmma_f32_16x16x32_bf16(false, fa, false, fb[3],
                                                         (short)0, acc[3], false, false);
        // stage 2: prefetch k0+64 (if any), then consume k0+32
        if (k0 + 64 < K) load_frag(arow, brow, N, k0 + 64, kg, fa, fb);
        acc[0] = __builtin_amdgcn_wmma_f32_16x16x32_bf16(false, ga, false, gb[0],
                                                         (short)0, acc[0], false, false);
        acc[1] = __builtin_amdgcn_wmma_f32_16x16x32_bf16(false, ga, false, gb[1],
                                                         (short)0, acc[1], false, false);
        acc[2] = __builtin_amdgcn_wmma_f32_16x16x32_bf16(false, ga, false, gb[2],
                                                         (short)0, acc[2], false, false);
        acc[3] = __builtin_amdgcn_wmma_f32_16x16x32_bf16(false, ga, false, gb[3],
                                                         (short)0, acc[3], false, false);
    }

    int rbase = (mT << 4) + ((lane >> 4) << 3);
#pragma unroll
    for (int t = 0; t < 4; ++t) {
        int col = (nB << 6) + (t << 4) + (lane & 15);
        float bs = bias ? bias[col] : 0.f;
#pragma unroll
        for (int r = 0; r < 8; ++r) {
            float v = acc[t][r] + bs;
            if (relu) v = fmaxf(v, 0.f);
            C[(size_t)(rbase + r) * N + col] = v;
        }
    }
}

// ---------------------------------------------------------------------------
// Wave-per-row dot with tiny output width J: out[row][j] = X[row]·Wj[j] + bj[j]
// (+ optional addend, e.g. anchors for pnt_coords).
// ---------------------------------------------------------------------------
template <int J>
__global__ void __launch_bounds__(256)
rowdot_kernel(const float* __restrict__ X, const float* __restrict__ Wj,
              const float* __restrict__ bj, const float* __restrict__ addend,
              float* __restrict__ out, int rows) {
    int wid  = (int)((blockIdx.x * blockDim.x + threadIdx.x) >> 5);
    int lane = threadIdx.x & 31;
    if (wid >= rows) return;
    const float* xr = X + (size_t)wid * HID;
    float s[J];
#pragma unroll
    for (int j = 0; j < J; ++j) s[j] = 0.f;
#pragma unroll
    for (int i = 0; i < 8; ++i) {
        float x = xr[lane + 32 * i];
#pragma unroll
        for (int j = 0; j < J; ++j) s[j] += x * Wj[j * HID + lane + 32 * i];
    }
#pragma unroll
    for (int j = 0; j < J; ++j)
        for (int mlt = 16; mlt >= 1; mlt >>= 1) s[j] += __shfl_xor(s[j], mlt, 32);
    if (lane == 0) {
#pragma unroll
        for (int j = 0; j < J; ++j) {
            float v = s[j] + bj[j];
            if (addend) v += addend[(size_t)wid * J + j];
            out[(size_t)wid * J + j] = v;
        }
    }
}

// ---------------------------------------------------------------------------
// Fused: logits(4) = Hh[row]·W2 + b2 ; softmax over 4 ; out = Σ_l w_l*roi[l].
// One block per row, 256 threads (one per channel).
// ---------------------------------------------------------------------------
__global__ void __launch_bounds__(256)
attn_mix_kernel(const float* __restrict__ Hh, const float* __restrict__ W2,
                const float* __restrict__ b2, const float* __restrict__ roi,
                float* __restrict__ outF) {
    int row = blockIdx.x;
    int t   = threadIdx.x;
    __shared__ float red[4][8];
    __shared__ float wgt[4];
    float x  = Hh[(size_t)row * HID + t];
    float l0 = x * W2[t], l1 = x * W2[HID + t];
    float l2 = x * W2[2 * HID + t], l3 = x * W2[3 * HID + t];
    for (int mlt = 16; mlt >= 1; mlt >>= 1) {
        l0 += __shfl_xor(l0, mlt, 32); l1 += __shfl_xor(l1, mlt, 32);
        l2 += __shfl_xor(l2, mlt, 32); l3 += __shfl_xor(l3, mlt, 32);
    }
    int wv = t >> 5;
    if ((t & 31) == 0) { red[0][wv] = l0; red[1][wv] = l1; red[2][wv] = l2; red[3][wv] = l3; }
    __syncthreads();
    if (t == 0) {
        float lg[4];
#pragma unroll
        for (int j = 0; j < 4; ++j) {
            float s = b2[j];
            for (int w = 0; w < 8; ++w) s += red[j][w];
            lg[j] = s;
        }
        float mx = fmaxf(fmaxf(lg[0], lg[1]), fmaxf(lg[2], lg[3]));
        float e0 = expf(lg[0] - mx), e1 = expf(lg[1] - mx);
        float e2 = expf(lg[2] - mx), e3 = expf(lg[3] - mx);
        float inv = 1.f / (e0 + e1 + e2 + e3);
        wgt[0] = e0 * inv; wgt[1] = e1 * inv; wgt[2] = e2 * inv; wgt[3] = e3 * inv;
    }
    __syncthreads();
    const float* rr = roi + (size_t)row * (4 * HID);
    outF[(size_t)row * HID + t] = wgt[0] * rr[t] + wgt[1] * rr[HID + t] +
                                  wgt[2] * rr[2 * HID + t] + wgt[3] * rr[3 * HID + t];
}

// ---------------------------------------------------------------------------
extern "C" void kernel_launch(void* const* d_in, const int* in_sizes, int n_in,
                              void* d_out, int out_size, void* d_ws, size_t ws_size,
                              hipStream_t stream) {
    (void)in_sizes; (void)n_in; (void)out_size; (void)ws_size;
    const float* feat[4]  = {(const float*)d_in[0], (const float*)d_in[1],
                             (const float*)d_in[2], (const float*)d_in[3]};
    const float* anchors  = (const float*)d_in[4];
    const float* dmlp_w1  = (const float*)d_in[5];
    const float* dmlp_b1  = (const float*)d_in[6];
    const float* dmlp_w2  = (const float*)d_in[7];
    const float* dmlp_b2  = (const float*)d_in[8];
    const float* reg_w1   = (const float*)d_in[9];
    const float* reg_b1   = (const float*)d_in[10];
    const float* reg_w2   = (const float*)d_in[11];
    const float* reg_b2   = (const float*)d_in[12];
    const float* reg_w3   = (const float*)d_in[13];
    const float* reg_b3   = (const float*)d_in[14];
    const float* cls_w    = (const float*)d_in[15];
    const float* cls_b    = (const float*)d_in[16];
    const float* la_w1    = (const float*)d_in[17];
    const float* la_b1    = (const float*)d_in[18];
    const float* la_w2    = (const float*)d_in[19];
    const float* la_b2    = (const float*)d_in[20];
    const float* ca_w1    = (const float*)d_in[21];
    const float* ca_b1    = (const float*)d_in[22];
    const float* ca_w2    = (const float*)d_in[23];
    const float* ca_b2    = (const float*)d_in[24];

    // ---- workspace layout (256B aligned regions) ----
    char*  base = (char*)d_ws;
    size_t cur  = 0;
    auto alloc = [&](size_t bytes) -> char* {
        char* p = base + cur;
        cur += (bytes + 255) & ~(size_t)255;
        return p;
    };
    float* roi  = (float*)alloc((size_t)ROWS * 1024 * 4);  // (rows, L*C) == af
    float* pre  = (float*)alloc((size_t)ROWS * HID * 4);   // sampled / h1
    float* hid  = (float*)alloc((size_t)ROWS * HID * 4);   // GEMM hidden
    float* regF = (float*)alloc((size_t)ROWS * HID * 4);
    float* clsF = (float*)alloc((size_t)ROWS * HID * 4);
    float* offb = (float*)alloc((size_t)ROWS * 2 * 4);
    unsigned short* wt_dmlp = (unsigned short*)alloc((size_t)4 * HID * HID * 2);
    unsigned short* wt_la   = (unsigned short*)alloc((size_t)1024 * HID * 2);
    unsigned short* wt_ca   = (unsigned short*)alloc((size_t)1024 * HID * 2);
    unsigned short* wt_r1   = (unsigned short*)alloc((size_t)HID * HID * 2);
    unsigned short* wt_r2   = (unsigned short*)alloc((size_t)HID * HID * 2);

    // ---- 1) weights -> bf16 transposed (K x N) ----
    for (int l = 0; l < 4; ++l)
        convert_wt_kernel<<<256, 256, 0, stream>>>(dmlp_w1 + (size_t)l * HID * HID,
                                                   wt_dmlp + (size_t)l * HID * HID,
                                                   HID, HID);
    convert_wt_kernel<<<1024, 256, 0, stream>>>(la_w1, wt_la, HID, 1024);
    convert_wt_kernel<<<1024, 256, 0, stream>>>(ca_w1, wt_ca, HID, 1024);
    convert_wt_kernel<<<256, 256, 0, stream>>>(reg_w1, wt_r1, HID, HID);
    convert_wt_kernel<<<256, 256, 0, stream>>>(reg_w2, wt_r2, HID, HID);

    // GEMM geometry: waves = (M/16)*(N/64) = 320*4 = 1280 -> 320 blocks x 128.
    const int gemmBlocks = 320;

    // ---- 2) per level: sample -> dmlp (WMMA) -> offsets -> re-sample -> roi
    const int dims[4] = {256, 128, 64, 32};
    for (int l = 0; l < 4; ++l) {
        float scale = (float)(dims[l] - 1) / 512.0f;  // (W-1)/(s*W), s*W==512
        sample_kernel<<<ROWS, 256, 0, stream>>>(feat[l], anchors, nullptr, pre,
                                                dims[l], scale, HID);
        gemm_bf16_wmma_kernel<<<gemmBlocks, 128, 0, stream>>>(
            pre, wt_dmlp + (size_t)l * HID * HID, dmlp_b1 + (size_t)l * HID,
            hid, ROWS, HID, HID, 1);
        rowdot_kernel<2><<<640, 256, 0, stream>>>(hid, dmlp_w2 + (size_t)l * 2 * HID,
                                                  dmlp_b2 + (size_t)l * 2, nullptr,
                                                  offb, ROWS);
        sample_kernel<<<ROWS, 256, 0, stream>>>(feat[l], anchors, offb,
                                                roi + (size_t)l * HID,
                                                dims[l], scale, 4 * HID);
    }

    // ---- 3) level-attention (WMMA K=1024) + softmax mix ----
    gemm_bf16_wmma_kernel<<<gemmBlocks, 128, 0, stream>>>(roi, wt_la, la_b1, hid,
                                                          ROWS, HID, 1024, 1);
    attn_mix_kernel<<<ROWS, 256, 0, stream>>>(hid, la_w2, la_b2, roi, regF);
    gemm_bf16_wmma_kernel<<<gemmBlocks, 128, 0, stream>>>(roi, wt_ca, ca_b1, hid,
                                                          ROWS, HID, 1024, 1);
    attn_mix_kernel<<<ROWS, 256, 0, stream>>>(hid, ca_w2, ca_b2, roi, clsF);

    // ---- 4) heads ----
    gemm_bf16_wmma_kernel<<<gemmBlocks, 128, 0, stream>>>(regF, wt_r1, reg_b1, pre,
                                                          ROWS, HID, HID, 1);
    gemm_bf16_wmma_kernel<<<gemmBlocks, 128, 0, stream>>>(pre, wt_r2, reg_b2, hid,
                                                          ROWS, HID, HID, 1);
    float* out_pnt = (float*)d_out;            // (rows, 2)
    float* out_cls = (float*)d_out + ROWS * 2; // (rows, 3)
    rowdot_kernel<2><<<640, 256, 0, stream>>>(hid, reg_w3, reg_b3, anchors,
                                              out_pnt, ROWS);
    rowdot_kernel<3><<<640, 256, 0, stream>>>(clsF, cls_w, cls_b, nullptr,
                                              out_cls, ROWS);
}
